// GAT_33569464385905
// MI455X (gfx1250) — matrix operand order repaired and checked by reading, not compiled, not run on hardware
//
#include <hip/hip_runtime.h>
#include <hip/hip_bf16.h>
#include <math.h>

// ---------------------------------------------------------------------------
// 2-layer GAT for MI455X (gfx1250, wave32).
//   L1 GEMM: bf16 WMMA (v_wmma_f32_16x16x32_bf16); W1 staged in LDS
//            pre-swizzled into B-fragment layout -> ds_load_b128 reads.
//   Edge softmax: ordered-uint atomic-max + hw f32 atomic-add (agent scope).
//   L1 aggregation: one wave per edge, shfl-broadcast alphas, coalesced atomics.
// ---------------------------------------------------------------------------

typedef __attribute__((ext_vector_type(16))) __bf16 v16bf;
typedef __attribute__((ext_vector_type(8)))  float  v8f;

#define NEG_SLOPE 0.2f

__device__ __forceinline__ float leaky(float x) { return x > 0.f ? x : NEG_SLOPE * x; }

// Order-preserving float->uint key (monotonic for all finite floats).
__device__ __forceinline__ unsigned fkey(float f) {
  unsigned u = __float_as_uint(f);
  return (u & 0x80000000u) ? ~u : (u | 0x80000000u);
}
__device__ __forceinline__ float funkey(unsigned k) {
  return (k & 0x80000000u) ? __uint_as_float(k ^ 0x80000000u) : __uint_as_float(~k);
}
__device__ __forceinline__ void atomAddF(float* p, float v) {
  __hip_atomic_fetch_add(p, v, __ATOMIC_RELAXED, __HIP_MEMORY_SCOPE_AGENT);
}
__device__ __forceinline__ void atomMaxU(unsigned* p, unsigned v) {
  __hip_atomic_fetch_max(p, v, __ATOMIC_RELAXED, __HIP_MEMORY_SCOPE_AGENT);
}

// --------------------------- K1: h1 = x @ W1 (WMMA) ------------------------
// Block = 256 threads = 8 waves; each wave produces a 16-row x 128-col strip.
// W1 (128x128) staged in LDS as bf16, PRE-SWIZZLED into this code's contiguous-k B-fragment
// register layout: record(kb, nt, lane) = 16 contiguous bf16 =
//   { W1[(kb*32 + (lane>>4)*16 + i)*128 + nt*16 + (lane&15)] : i = 0..15 }
// so each B fragment is one aligned 32-byte LDS vector load.
__global__ __launch_bounds__(256) void gat_gemm1_wmma(
    const float* __restrict__ x, const float* __restrict__ W1,
    float* __restrict__ h1, int N)
{
  __shared__ __attribute__((aligned(32))) __bf16 blds[128 * 128];  // 32 KB
  const int tid = threadIdx.x;
  // Swizzled stage: idx = ((kb*8 + nt)*32 + lane_e)*16 + i
  for (int idx = tid; idx < 128 * 128; idx += 256) {
    const int i      = idx & 15;
    const int lane_e = (idx >> 4) & 31;
    const int nt     = (idx >> 9) & 7;
    const int kb     = idx >> 12;
    const int k   = kb * 32 + (lane_e >> 4) * 16 + i;
    const int col = nt * 16 + (lane_e & 15);
    blds[idx] = (__bf16)W1[k * 128 + col];
  }
  __syncthreads();

  const int lane = tid & 31, wave = tid >> 5;
  const int hs = lane >> 4;   // lane half-select (0: lanes 0-15, 1: lanes 16-31)
  const int lr = lane & 15;
  const long m0 = (long)blockIdx.x * 128 + (long)wave * 16;

  long arow = m0 + lr;                       // A-frag: row per lane
  if (arow >= N) arow = N - 1;               // clamp: keep EXEC all-ones for WMMA
  const float* __restrict__ xr = x + (size_t)arow * 128;

  v8f acc[8];
#pragma unroll
  for (int i = 0; i < 8; ++i) acc[i] = (v8f){0, 0, 0, 0, 0, 0, 0, 0};

#pragma unroll
  for (int kb = 0; kb < 4; ++kb) {           // K = 128 in blocks of 32
    // A fragment (16x32 bf16): elems 0..15 -> K = kb*32 + hs*16 + i (matches B record)
    v16bf a;
    const int k0 = kb * 32 + hs * 16;
#pragma unroll
    for (int i = 0; i < 8; ++i) a[i] = (__bf16)xr[k0 + i];
#pragma unroll
    for (int i = 0; i < 8; ++i) a[8 + i] = (__bf16)xr[k0 + 8 + i];

#pragma unroll
    for (int nt = 0; nt < 8; ++nt) {
      // One 32-byte vector LDS read per B fragment (ds_load_b128 x2).
      const v16bf b = *(const v16bf*)(blds + ((((kb << 3) + nt) << 5) + lane) * 16);
      acc[nt] = __builtin_amdgcn_wmma_f32_16x16x32_bf16(
          false, a, false, b, (short)0, acc[nt], false, false);
    }
  }

  // C/D layout: element v of v8f -> row m0 + hs*8 + v, col nt*16 + lr
#pragma unroll
  for (int nt = 0; nt < 8; ++nt) {
    const int col = nt * 16 + lr;
#pragma unroll
    for (int v = 0; v < 8; ++v) {
      const long r = m0 + hs * 8 + v;
      if (r < N) h1[(size_t)r * 128 + col] = acc[nt][v];
    }
  }
}

// ----------------- K2: el/er per (node, head); init buffers ----------------
__global__ void gat_node_prep1(
    const float* __restrict__ h1, const float* __restrict__ al,
    const float* __restrict__ ar, float* __restrict__ el, float* __restrict__ er,
    unsigned* __restrict__ maxkey, float* __restrict__ sumz,
    float* __restrict__ out1, int N)
{
  const int t = blockIdx.x * blockDim.x + threadIdx.x;
  if (t >= N * 8) return;
  const int n = t >> 3, h = t & 7;
  const float4* hp = (const float4*)(h1 + (size_t)n * 128 + h * 16);
  const float4* ap = (const float4*)(al + h * 16);
  const float4* rp = (const float4*)(ar + h * 16);
  float sl = 0.f, sr = 0.f;
#pragma unroll
  for (int i = 0; i < 4; ++i) {
    const float4 hv = hp[i], av = ap[i], rv = rp[i];
    sl += hv.x * av.x + hv.y * av.y + hv.z * av.z + hv.w * av.w;
    sr += hv.x * rv.x + hv.y * rv.y + hv.z * rv.z + hv.w * rv.w;
  }
  el[t] = sl;
  er[t] = sr;
  maxkey[t] = 0u;   // fkey^-1(0) < fkey(any finite)
  sumz[t] = 0.f;
  float4* op = (float4*)(out1 + (size_t)n * 128 + h * 16);
  const float4 z = {0.f, 0.f, 0.f, 0.f};
#pragma unroll
  for (int i = 0; i < 4; ++i) op[i] = z;
}

// --------------------- K3: segment max over edges (L1) ---------------------
__global__ void gat_edge_max1(
    const int* __restrict__ src, const int* __restrict__ dst,
    const float* __restrict__ el, const float* __restrict__ er,
    unsigned* __restrict__ maxkey, int E)
{
  const int t = blockIdx.x * blockDim.x + threadIdx.x;
  if (t >= E) return;
  const int s = src[t], d = dst[t];
  const float4* lp = (const float4*)(el + (size_t)s * 8);
  const float4* rp = (const float4*)(er + (size_t)d * 8);
  const float4 l0 = lp[0], l1 = lp[1], r0 = rp[0], r1 = rp[1];
  const float ev[8] = {l0.x + r0.x, l0.y + r0.y, l0.z + r0.z, l0.w + r0.w,
                       l1.x + r1.x, l1.y + r1.y, l1.z + r1.z, l1.w + r1.w};
#pragma unroll
  for (int h = 0; h < 8; ++h)
    atomMaxU(&maxkey[(size_t)d * 8 + h], fkey(leaky(ev[h])));
}

// --------------------- K4: segment sum of exp(e - m) (L1) ------------------
__global__ void gat_edge_sum1(
    const int* __restrict__ src, const int* __restrict__ dst,
    const float* __restrict__ el, const float* __restrict__ er,
    const unsigned* __restrict__ maxkey, float* __restrict__ sumz, int E)
{
  const int t = blockIdx.x * blockDim.x + threadIdx.x;
  if (t >= E) return;
  const int s = src[t], d = dst[t];
  const float4* lp = (const float4*)(el + (size_t)s * 8);
  const float4* rp = (const float4*)(er + (size_t)d * 8);
  const float4 l0 = lp[0], l1 = lp[1], r0 = rp[0], r1 = rp[1];
  const float ev[8] = {l0.x + r0.x, l0.y + r0.y, l0.z + r0.z, l0.w + r0.w,
                       l1.x + r1.x, l1.y + r1.y, l1.z + r1.z, l1.w + r1.w};
#pragma unroll
  for (int h = 0; h < 8; ++h) {
    const float m = funkey(maxkey[(size_t)d * 8 + h]);
    atomAddF(&sumz[(size_t)d * 8 + h], expf(leaky(ev[h]) - m));
  }
}

// ------------- K5: out1[dst] += alpha * h1[src]  (one wave / edge) ---------
__global__ __launch_bounds__(256) void gat_edge_agg1(
    const int* __restrict__ src, const int* __restrict__ dst,
    const float* __restrict__ el, const float* __restrict__ er,
    const unsigned* __restrict__ maxkey, const float* __restrict__ sumz,
    const float* __restrict__ h1, float* __restrict__ out1, int E)
{
  const int wave = threadIdx.x >> 5, lane = threadIdx.x & 31;
  const long e = (long)blockIdx.x * 8 + wave;
  if (e >= E) return;
  const int s = src[e], d = dst[e];
  float alpha = 0.f;
  if (lane < 8) {  // lanes 0..7 own heads 0..7
    const float ev = leaky(el[(size_t)s * 8 + lane] + er[(size_t)d * 8 + lane]);
    const float m  = funkey(maxkey[(size_t)d * 8 + lane]);
    alpha = expf(ev - m) / sumz[(size_t)d * 8 + lane];
  }
  const float* __restrict__ hsrc = h1 + (size_t)s * 128;
  float* __restrict__ od = out1 + (size_t)d * 128;
#pragma unroll
  for (int j = 0; j < 4; ++j) {
    const int f = lane + 32 * j;              // feature; head = f >> 4
    const float a = __shfl(alpha, f >> 4, 32);
    atomAddF(od + f, a * hsrc[f]);
  }
}

// ---- K6: bias + ELU in place; h2 = helu @ W2; scalar attn coefs (L2) ------
__global__ __launch_bounds__(256) void gat_node_prep2(
    float* __restrict__ helu, const float* __restrict__ b1,
    const float* __restrict__ W2, const float* __restrict__ al2,
    const float* __restrict__ ar2, float* __restrict__ h2,
    float* __restrict__ el2, float* __restrict__ er2,
    unsigned* __restrict__ maxkey2, float* __restrict__ sum2,
    float* __restrict__ dout, int N)
{
  const int wave = threadIdx.x >> 5, lane = threadIdx.x & 31;
  const long n = (long)blockIdx.x * 8 + wave;
  if (n >= N) return;
  float4* hp = (float4*)(helu + (size_t)n * 128);
  const float4 bv = ((const float4*)b1)[lane];
  const float4 wv = ((const float4*)W2)[lane];
  float4 hv = hp[lane];
  hv.x += bv.x; hv.y += bv.y; hv.z += bv.z; hv.w += bv.w;
  hv.x = hv.x > 0.f ? hv.x : expm1f(hv.x);
  hv.y = hv.y > 0.f ? hv.y : expm1f(hv.y);
  hv.z = hv.z > 0.f ? hv.z : expm1f(hv.z);
  hv.w = hv.w > 0.f ? hv.w : expm1f(hv.w);
  hp[lane] = hv;
  float p = hv.x * wv.x + hv.y * wv.y + hv.z * wv.z + hv.w * wv.w;
#pragma unroll
  for (int o = 16; o >= 1; o >>= 1) p += __shfl_down(p, o, 32);
  if (lane == 0) {
    h2[n] = p;
    el2[n] = p * al2[0];
    er2[n] = p * ar2[0];
    maxkey2[n] = 0u;
    sum2[n] = 0.f;
    dout[n] = 0.f;
  }
}

// ----------------------- K7/K8/K9: layer-2 edge passes ---------------------
__global__ void gat_edge_max2(
    const int* __restrict__ src, const int* __restrict__ dst,
    const float* __restrict__ el2, const float* __restrict__ er2,
    unsigned* __restrict__ maxkey2, int E)
{
  const int t = blockIdx.x * blockDim.x + threadIdx.x;
  if (t >= E) return;
  atomMaxU(&maxkey2[dst[t]], fkey(leaky(el2[src[t]] + er2[dst[t]])));
}

__global__ void gat_edge_sum2(
    const int* __restrict__ src, const int* __restrict__ dst,
    const float* __restrict__ el2, const float* __restrict__ er2,
    const unsigned* __restrict__ maxkey2, float* __restrict__ sum2, int E)
{
  const int t = blockIdx.x * blockDim.x + threadIdx.x;
  if (t >= E) return;
  const int s = src[t], d = dst[t];
  const float ev = leaky(el2[s] + er2[d]);
  atomAddF(&sum2[d], expf(ev - funkey(maxkey2[d])));
}

__global__ void gat_edge_agg2(
    const int* __restrict__ src, const int* __restrict__ dst,
    const float* __restrict__ el2, const float* __restrict__ er2,
    const unsigned* __restrict__ maxkey2, const float* __restrict__ sum2,
    const float* __restrict__ h2, float* __restrict__ dout, int E)
{
  const int t = blockIdx.x * blockDim.x + threadIdx.x;
  if (t >= E) return;
  const int s = src[t], d = dst[t];
  const float ev = leaky(el2[s] + er2[d]);
  const float alpha = expf(ev - funkey(maxkey2[d])) / sum2[d];
  atomAddF(&dout[d], alpha * h2[s]);
}

__global__ void gat_bias2(float* __restrict__ dout, const float* __restrict__ b2, int N) {
  const int t = blockIdx.x * blockDim.x + threadIdx.x;
  if (t < N) dout[t] += b2[0];
}

// ---------------------------------------------------------------------------
extern "C" void kernel_launch(void* const* d_in, const int* in_sizes, int n_in,
                              void* d_out, int out_size, void* d_ws, size_t ws_size,
                              hipStream_t stream)
{
  const float* x   = (const float*)d_in[0];
  const int*   src = (const int*)d_in[1];
  const int*   dst = (const int*)d_in[2];
  const float* W1  = (const float*)d_in[3];
  const float* al1 = (const float*)d_in[4];
  const float* ar1 = (const float*)d_in[5];
  const float* b1  = (const float*)d_in[6];
  const float* W2  = (const float*)d_in[7];
  const float* al2 = (const float*)d_in[8];
  const float* ar2 = (const float*)d_in[9];
  const float* b2  = (const float*)d_in[10];
  const int N = in_sizes[0] / 128;
  const int E = in_sizes[1];
  float* dout = (float*)d_out;

  // workspace partition (~58.6 MB for N=50000)
  float* ws = (float*)d_ws;
  size_t o = 0;
  float*    h1   = ws + o; o += (size_t)N * 128;
  float*    out1 = ws + o; o += (size_t)N * 128;
  float*    el1  = ws + o; o += (size_t)N * 8;
  float*    er1  = ws + o; o += (size_t)N * 8;
  float*    sum1 = ws + o; o += (size_t)N * 8;
  unsigned* mk1  = (unsigned*)(ws + o); o += (size_t)N * 8;
  float*    h2   = ws + o; o += (size_t)N;
  float*    el2  = ws + o; o += (size_t)N;
  float*    er2  = ws + o; o += (size_t)N;
  float*    sum2 = ws + o; o += (size_t)N;
  unsigned* mk2  = (unsigned*)(ws + o); o += (size_t)N;

  gat_gemm1_wmma<<<(N + 127) / 128, 256, 0, stream>>>(x, W1, h1, N);
  gat_node_prep1<<<(N * 8 + 255) / 256, 256, 0, stream>>>(h1, al1, ar1, el1, er1, mk1, sum1, out1, N);
  gat_edge_max1<<<(E + 255) / 256, 256, 0, stream>>>(src, dst, el1, er1, mk1, E);
  gat_edge_sum1<<<(E + 255) / 256, 256, 0, stream>>>(src, dst, el1, er1, mk1, sum1, E);
  gat_edge_agg1<<<(E + 7) / 8, 256, 0, stream>>>(src, dst, el1, er1, mk1, sum1, h1, out1, E);
  gat_node_prep2<<<(N + 7) / 8, 256, 0, stream>>>(out1, b1, W2, al2, ar2, h2, el2, er2, mk2, sum2, dout, N);
  gat_edge_max2<<<(E + 255) / 256, 256, 0, stream>>>(src, dst, el2, er2, mk2, E);
  gat_edge_sum2<<<(E + 255) / 256, 256, 0, stream>>>(src, dst, el2, er2, mk2, sum2, E);
  gat_edge_agg2<<<(E + 255) / 256, 256, 0, stream>>>(src, dst, el2, er2, mk2, sum2, h2, dout, E);
  gat_bias2<<<(N + 255) / 256, 256, 0, stream>>>(dout, b2, N);
}
